// Qwen1Block_23742579212769
// MI455X (gfx1250) — matrix operand order repaired and checked by reading, not compiled, not run on hardware
//
#include <hip/hip_runtime.h>

// ---------------------------------------------------------------------------
// Qwen block for MI455X (gfx1250): bf16 WMMA GEMMs + flash attention,
// with Tensor Data Mover (TDM) double-buffered LDS staging everywhere.
// Shapes: B=1, S=2048, H=4096, NH=32, D=128, FF=11008, 3H=12288.
// ---------------------------------------------------------------------------

typedef unsigned short u16;
typedef unsigned long long u64;
typedef __attribute__((ext_vector_type(16))) __bf16 bf16x16;
typedef __attribute__((ext_vector_type(8)))  float  f32x8;
typedef __attribute__((ext_vector_type(4)))  unsigned int u32x4;
typedef __attribute__((ext_vector_type(8)))  int    i32x8;
typedef __attribute__((ext_vector_type(4)))  int    i32x4;

#define HAVE_TDM __has_builtin(__builtin_amdgcn_tensor_load_to_lds)

union Frag {
    uint4   u4[2];
    u16     s[16];
    bf16x16 v;
};

__device__ __forceinline__ u16 f2bf(float f) {
    unsigned u = __float_as_uint(f);
    u += 0x7FFFu + ((u >> 16) & 1u);   // round-to-nearest-even
    return (u16)(u >> 16);
}
__device__ __forceinline__ float bf2f(u16 h) {
    return __uint_as_float(((unsigned)h) << 16);
}
__device__ __forceinline__ f32x8 zf8() {
    f32x8 z = {0.f, 0.f, 0.f, 0.f, 0.f, 0.f, 0.f, 0.f};
    return z;
}
__device__ __forceinline__ f32x8 wmma_bf16(const Frag& a, const Frag& b, f32x8 c) {
    // emits v_wmma_f32_16x16x32_bf16
    return __builtin_amdgcn_wmma_f32_16x16x32_bf16(
        false, a.v, false, b.v, (short)0, c, false, false);
}
// Low 32 bits of a flat LDS pointer are the LDS byte offset (aperture in [63:32]).
__device__ __forceinline__ unsigned lds_off32(const void* p) {
    return (unsigned)(u64)p;
}

#if HAVE_TDM
// D# group1: data_size=2B, pad_enable; tile tile0 x tile1 elements,
// LDS padding: pad_interval enc (2^(e+1) DWORDs), pad_amount enc (e+1 DWORDs).
__device__ __forceinline__ i32x8 tdm_g1(unsigned dim0, unsigned dim1, unsigned stride0,
                                        unsigned tile0, unsigned tile1,
                                        unsigned pad_int_enc, unsigned pad_amt_enc) {
    i32x8 g1;
    g1[0] = (int)((1u << 16) | (1u << 20) | (pad_int_enc << 22) | (pad_amt_enc << 25));
    g1[1] = (int)((dim0 & 0xFFFFu) << 16);
    g1[2] = (int)((dim0 >> 16) | ((dim1 & 0xFFFFu) << 16));
    g1[3] = (int)((dim1 >> 16) | (tile0 << 16));   // tile_dim0 [127:112]
    g1[4] = (int)tile1;                            // tile_dim1 [143:128], tile_dim2=0
    g1[5] = (int)stride0;                          // tensor_dim0_stride[31:0]
    g1[6] = 0;
    g1[7] = 0;
    return g1;
}
__device__ __forceinline__ void tdm_load(unsigned lds, u64 ga, const i32x8& g1) {
    u32x4 g0;
    g0.x = 1u;                                        // count=1 (user descriptor)
    g0.y = lds;                                       // LDS byte address
    g0.z = (unsigned)ga;                              // global_addr[31:0]
    g0.w = (unsigned)((ga >> 32) & 0x1FFFFFFu) | (2u << 30);  // [56:32] | type=2
    const i32x4 z4 = {0, 0, 0, 0};
#if defined(__clang_major__) && __clang_major__ >= 23
    const i32x8 z8 = {0, 0, 0, 0, 0, 0, 0, 0};
    __builtin_amdgcn_tensor_load_to_lds(g0, g1, z4, z4, z8, 0);
#else
    __builtin_amdgcn_tensor_load_to_lds(g0, g1, z4, z4, 0);
#endif
}
#endif

// ---------------------------------------------------------------------------
// fp32 [R][C] -> bf16 transposed [C][R]  (weights, once per launch)
// 64x64 tiles through LDS; coalesced reads and writes.
// ---------------------------------------------------------------------------
__global__ __launch_bounds__(256) void castT_k(const float* __restrict__ in, int ldi,
                                               u16* __restrict__ out, int ldo) {
    __shared__ u16 t[64][72];
    const int rb = blockIdx.x * 64, cb = blockIdx.y * 64;
    const int r0 = threadIdx.x >> 2, c0 = (threadIdx.x & 3) * 16;
    const float* src = in + (size_t)(rb + r0) * ldi + cb + c0;
    #pragma unroll
    for (int j = 0; j < 16; ++j) t[c0 + j][r0] = f2bf(src[j]);
    __syncthreads();
    const int cc = threadIdx.x >> 2, rr = (threadIdx.x & 3) * 16;
    u16* dst = out + (size_t)(cb + cc) * ldo + rb + rr;
    *(uint4*)(dst)     = *(const uint4*)&t[cc][rr];
    *(uint4*)(dst + 8) = *(const uint4*)&t[cc][rr + 8];
}

// ---------------------------------------------------------------------------
// bf16 [R][C] -> bf16 transposed [C][R], batched over z (per attention head)
// ---------------------------------------------------------------------------
__global__ __launch_bounds__(256) void transpose16_k(const u16* __restrict__ in, int ldi,
                                                     size_t in_zs,
                                                     u16* __restrict__ out, int ldo,
                                                     size_t out_zs) {
    __shared__ u16 t[64][72];
    const u16* inb = in + (size_t)blockIdx.z * in_zs;
    u16* outb = out + (size_t)blockIdx.z * out_zs;
    const int rb = blockIdx.x * 64, cb = blockIdx.y * 64;
    const int r0 = threadIdx.x >> 2, c0 = (threadIdx.x & 3) * 16;
    const u16* src = inb + (size_t)(rb + r0) * ldi + cb + c0;
    Frag u;
    u.u4[0] = *(const uint4*)(src);
    u.u4[1] = *(const uint4*)(src + 8);
    #pragma unroll
    for (int j = 0; j < 16; ++j) t[c0 + j][r0] = u.s[j];
    __syncthreads();
    const int cc = threadIdx.x >> 2, rr = (threadIdx.x & 3) * 16;
    u16* dst = outb + (size_t)(cb + cc) * ldo + rb + rr;
    *(uint4*)(dst)     = *(const uint4*)&t[cc][rr];
    *(uint4*)(dst + 8) = *(const uint4*)&t[cc][rr + 8];
}

// ---------------------------------------------------------------------------
// RMSNorm: one 256-thread block (8 waves) per row, bf16 output
// ---------------------------------------------------------------------------
__global__ __launch_bounds__(256) void rmsnorm_k(const float* __restrict__ x,
                                                 const float* __restrict__ w,
                                                 u16* __restrict__ out, int H) {
    const int row = blockIdx.x;
    const float* xr = x + (size_t)row * H;
    float ss = 0.f;
    for (int i = threadIdx.x; i < H; i += 256) { float v = xr[i]; ss += v * v; }
    #pragma unroll
    for (int m = 16; m >= 1; m >>= 1) ss += __shfl_xor(ss, m, 32);
    __shared__ float red[8];
    if ((threadIdx.x & 31) == 0) red[threadIdx.x >> 5] = ss;
    __syncthreads();
    float tot = 0.f;
    #pragma unroll
    for (int i = 0; i < 8; ++i) tot += red[i];
    const float rs = rsqrtf(tot / (float)H + 1e-6f);
    for (int i = threadIdx.x; i < H; i += 256)
        out[(size_t)row * H + i] = f2bf(xr[i] * rs * w[i]);
}

// ---------------------------------------------------------------------------
// bf16 WMMA GEMM: C[M,N] = A[M,K] * W[K,N] (+bias) (+residual), W given as
// WT[N][K]. BM=128 BN=128 BK=32; 8 waves in 2(M)x4(N), 4x2 wmma tiles each.
// Both A and B tiles are row-major 32x128 copies -> TDM double-buffered:
// wave0 issues next tile's two tensor_load_to_lds, then s_wait_tensorcnt<=2
// (TDM completes in order), so DMA of tile i+1 overlaps WMMA of tile i.
// ---------------------------------------------------------------------------
template <bool OUT_BF16, bool HAS_BIAS, bool HAS_RES>
__global__ __launch_bounds__(256) void gemm_bf16_k(const u16* __restrict__ A,
                                                   const u16* __restrict__ WT,
                                                   const float* __restrict__ bias,
                                                   const float* __restrict__ res,
                                                   void* __restrict__ outp,
                                                   int M, int N, int K) {
    __shared__ u16 As[2][128 * 40];
    __shared__ u16 Bs[2][128 * 40];

    const int tid  = threadIdx.x;
    const int lane = tid & 31;
    const int lh   = lane & 15;
    const int hi   = lane >> 4;            // 0 | 1 (lane half)
    const int w    = tid >> 5;             // wave 0..7
    const int mw   = w >> 2;               // 0..1
    const int nw   = w & 3;                // 0..3
    const int mb   = blockIdx.y * 128;
    const int nb   = blockIdx.x * 128;

    f32x8 acc[4][2];
    #pragma unroll
    for (int i = 0; i < 4; ++i)
        #pragma unroll
        for (int j = 0; j < 2; ++j) acc[i][j] = zf8();

    const int T = K >> 5;

#if HAVE_TDM
    const i32x8 g1a = tdm_g1((unsigned)K, (unsigned)M, (unsigned)K, 32u, 128u, 3u, 3u);
    const i32x8 g1b = tdm_g1((unsigned)K, (unsigned)N, (unsigned)K, 32u, 128u, 3u, 3u);
    const unsigned la[2] = {lds_off32(&As[0][0]), lds_off32(&As[1][0])};
    const unsigned lb[2] = {lds_off32(&Bs[0][0]), lds_off32(&Bs[1][0])};
    const u64 abase = (u64)(const void*)(A  + (size_t)mb * K);
    const u64 bbase = (u64)(const void*)(WT + (size_t)nb * K);
    if (tid < 32) {                       // prologue: tile 0
        tdm_load(la[0], abase, g1a);
        tdm_load(lb[0], bbase, g1b);
    }
#else
    const int ra = tid >> 1, ca = (tid & 1) * 16;   // manual staging map
#endif

    for (int it = 0; it < T; ++it) {
        const int buf = it & 1;
#if HAVE_TDM
        if (tid < 32) {
            if (it + 1 < T) {             // issue next tile into the other buffer
                const u64 koff = (u64)(it + 1) * 64u;   // 32 elems * 2B
                tdm_load(la[buf ^ 1], abase + koff, g1a);
                tdm_load(lb[buf ^ 1], bbase + koff, g1b);
                __builtin_amdgcn_s_wait_tensorcnt(2);   // current tile complete
            } else {
                __builtin_amdgcn_s_wait_tensorcnt(0);
            }
        }
#else
        {   // manual staging: A[M][K] tile and WT[N][K] tile, both row-major
            const int kt = it * 32;
            const u16* ag = A + (size_t)(mb + ra) * K + kt + ca;
            *(uint4*)&As[buf][ra * 40 + ca]     = *(const uint4*)(ag);
            *(uint4*)&As[buf][ra * 40 + ca + 8] = *(const uint4*)(ag + 8);
            const u16* bg = WT + (size_t)(nb + ra) * K + kt + ca;
            *(uint4*)&Bs[buf][ra * 40 + ca]     = *(const uint4*)(bg);
            *(uint4*)&Bs[buf][ra * 40 + ca + 8] = *(const uint4*)(bg + 8);
        }
#endif
        __syncthreads();

        // --- fragments ---
        Frag afr[4], bfr[2];
        #pragma unroll
        for (int mt = 0; mt < 4; ++mt) {
            const u16* ap = &As[buf][(mw * 64 + mt * 16 + lh) * 40 + (hi ? 8 : 0)];
            afr[mt].u4[0] = *(const uint4*)(ap);
            afr[mt].u4[1] = *(const uint4*)(ap + 16);
        }
        #pragma unroll
        for (int nt = 0; nt < 2; ++nt) {
            const u16* bp = &Bs[buf][(nw * 32 + nt * 16 + lh) * 40 + (hi ? 8 : 0)];
            bfr[nt].u4[0] = *(const uint4*)(bp);
            bfr[nt].u4[1] = *(const uint4*)(bp + 16);
        }
        #pragma unroll
        for (int mt = 0; mt < 4; ++mt)
            #pragma unroll
            for (int nt = 0; nt < 2; ++nt)
                acc[mt][nt] = wmma_bf16(afr[mt], bfr[nt], acc[mt][nt]);
        __syncthreads();
    }

    // --- epilogue ---
    #pragma unroll
    for (int mt = 0; mt < 4; ++mt) {
        #pragma unroll
        for (int nt = 0; nt < 2; ++nt) {
            const int col = nb + nw * 32 + nt * 16 + lh;
            const float bv = HAS_BIAS ? bias[col] : 0.f;
            #pragma unroll
            for (int r = 0; r < 8; ++r) {
                const int row = mb + mw * 64 + mt * 16 + r + hi * 8;
                const size_t idx = (size_t)row * N + col;
                float v = acc[mt][nt][r] + bv;
                if (HAS_RES) v += res[idx];
                if (OUT_BF16) ((u16*)outp)[idx] = f2bf(v);
                else          ((float*)outp)[idx] = v;
            }
        }
    }
}

// ---------------------------------------------------------------------------
// RoPE: rotate-half over D=128 in fp32; reads bf16 qkv, writes bf16 q,k
// ---------------------------------------------------------------------------
__global__ __launch_bounds__(256) void rope_k(const u16* __restrict__ qkv,
                                              const float* __restrict__ cosb,
                                              const float* __restrict__ sinb,
                                              u16* __restrict__ qo,
                                              u16* __restrict__ ko) {
    const int g = blockIdx.x * 256 + threadIdx.x;   // S*NH*64 threads
    const int d = g & 63;
    const int h = (g >> 6) & 31;
    const int s = g >> 11;
    const u16* row = qkv + (size_t)s * 12288 + h * 128;
    const float q1 = bf2f(row[d]),        q2 = bf2f(row[d + 64]);
    const float k1 = bf2f(row[4096 + d]), k2 = bf2f(row[4096 + d + 64]);
    const float c1 = cosb[s * 128 + d], c2 = cosb[s * 128 + d + 64];
    const float s1 = sinb[s * 128 + d], s2 = sinb[s * 128 + d + 64];
    const size_t ob = (size_t)s * 4096 + h * 128;
    qo[ob + d]      = f2bf(q1 * c1 - q2 * s1);
    qo[ob + d + 64] = f2bf(q2 * c2 + q1 * s2);
    ko[ob + d]      = f2bf(k1 * c1 - k2 * s1);
    ko[ob + d + 64] = f2bf(k2 * c2 + k1 * s2);
}

// ---------------------------------------------------------------------------
// Flash attention (causal). Grid: (S/128, NH). 8 waves; wave owns 16 q rows.
// KV blocks of 32. Q is register-resident; K and V^T tiles are both row-major
// TDM copies, double-buffered (wave0: issue 2 loads for block b+1, then
// s_wait_tensorcnt<=2 for block b). The inner loop touches no global memory.
// K tile: 32 rows x 128 dims, rows 256B + 16B pad (stride 272B -> lanes hit
// 16 distinct banks). V^T tile: 128 rows x 32 keys, rows 64B + 16B pad.
// P goes through a per-wave LDS round-trip (DS ops are in-order per wave).
// ---------------------------------------------------------------------------
__global__ __launch_bounds__(256) void attn_k(const u16* __restrict__ Q, int ldq,
                                              const u16* __restrict__ Km, int ldk,
                                              const u16* __restrict__ VTg, int S,
                                              u16* __restrict__ O, int ldo) {
    __shared__ u16 Ks[2][32 * 136];     // Ks[buf][key][dim], stride 136 u16 = 272B
    __shared__ u16 VT[2][128 * 40];     // VT[buf][dim][key]
    __shared__ u16 Ps[8 * 16 * 40];     // per-wave P tiles [16 rows][32 keys]

    const float SCALE = 0.08838834764831845f;   // 1/sqrt(128)
    const int tid  = threadIdx.x;
    const int lane = tid & 31;
    const int lh   = lane & 15;
    const int hi   = lane >> 4;
    const int w    = tid >> 5;
    const int qb   = blockIdx.x * 128;
    const int hoff = blockIdx.y * 128;
    const int wq   = qb + w * 16;       // wave's first q row
    const int pbase = w * 16 * 40;

    // preload Q fragments (16 rows x 128 dims -> 4 K-chunks of 32)
    Frag qf[4];
    const u16* qrow = Q + (size_t)(wq + lh) * ldq + hoff;
    #pragma unroll
    for (int c = 0; c < 4; ++c) {
        const int c0 = c * 32 + (hi ? 8 : 0);
        qf[c].u4[0] = *(const uint4*)(qrow + c0);
        qf[c].u4[1] = *(const uint4*)(qrow + c0 + 16);
    }

    f32x8 o[8];
    #pragma unroll
    for (int t = 0; t < 8; ++t) o[t] = zf8();
    float m_run[8], l_run[8];
    #pragma unroll
    for (int r = 0; r < 8; ++r) { m_run[r] = -1e30f; l_run[r] = 0.f; }

    const int nblk = qb / 32 + 4;       // causal: keys up to qb+127
    const u16* vbase = VTg + (size_t)hoff * S;   // head slab: [128 dims][S keys]
    const u16* kbase = Km + hoff;                // rows of [S][ldk], head slice

#if HAVE_TDM
    const i32x8 g1v = tdm_g1((unsigned)S, 128u, (unsigned)S, 32u, 128u, 3u, 3u);
    const i32x8 g1k = tdm_g1(128u, (unsigned)S, (unsigned)ldk, 128u, 32u, 5u, 3u);
    const unsigned lv[2] = {lds_off32(&VT[0][0]), lds_off32(&VT[1][0])};
    const unsigned lk[2] = {lds_off32(&Ks[0][0]), lds_off32(&Ks[1][0])};
    const u64 vga = (u64)(const void*)vbase;
    const u64 kga = (u64)(const void*)kbase;
    const u64 kbytes = (u64)32 * (unsigned)ldk * 2u;   // 32 key rows per block
    if (tid < 32) {                     // prologue: block 0
        tdm_load(lv[0], vga, g1v);
        tdm_load(lk[0], kga, g1k);
    }
#else
    const int rd = tid >> 1, cd = (tid & 1) * 16;      // V staging map
    const int kk = tid >> 3, kd = (tid & 7) * 16;      // K staging map
#endif

    for (int b = 0; b < nblk; ++b) {
        const int kb = b * 32;
        const int buf = b & 1;
#if HAVE_TDM
        if (tid < 32) {
            if (b + 1 < nblk) {
                tdm_load(lv[buf ^ 1], vga + (u64)(b + 1) * 64u, g1v);
                tdm_load(lk[buf ^ 1], kga + (u64)(b + 1) * kbytes, g1k);
                __builtin_amdgcn_s_wait_tensorcnt(2);
            } else {
                __builtin_amdgcn_s_wait_tensorcnt(0);
            }
        }
#else
        {   // manual staging, both tiles row-major contiguous copies
            const u16* vg = vbase + (size_t)rd * S + kb + cd;
            *(uint4*)&VT[buf][rd * 40 + cd]     = *(const uint4*)(vg);
            *(uint4*)&VT[buf][rd * 40 + cd + 8] = *(const uint4*)(vg + 8);
            const u16* kg = kbase + (size_t)(kb + kk) * ldk + kd;
            *(uint4*)&Ks[buf][kk * 136 + kd]     = *(const uint4*)(kg);
            *(uint4*)&Ks[buf][kk * 136 + kd + 8] = *(const uint4*)(kg + 8);
        }
#endif
        __syncthreads();

        if (kb <= wq + 15) {            // wave-uniform: EXEC stays full
            // ---- K fragments from LDS, then the WMMA chain ----
            const u16* kp0 = &Ks[buf][lh * 136];
            const u16* kp1 = &Ks[buf][(16 + lh) * 136];
            Frag kf[8];
            #pragma unroll
            for (int c = 0; c < 4; ++c) {
                const int off = c * 32 + hi * 8;
                kf[c].u4[0]     = *(const uint4*)(kp0 + off);
                kf[c].u4[1]     = *(const uint4*)(kp0 + off + 16);
                kf[4 + c].u4[0] = *(const uint4*)(kp1 + off);
                kf[4 + c].u4[1] = *(const uint4*)(kp1 + off + 16);
            }
            f32x8 st[2] = {zf8(), zf8()};
            #pragma unroll
            for (int c = 0; c < 4; ++c) {
                st[0] = wmma_bf16(qf[c], kf[c],     st[0]);
                st[1] = wmma_bf16(qf[c], kf[4 + c], st[1]);
            }
            // ---- online softmax over the 32 keys ----
            float al[8];
            #pragma unroll
            for (int r = 0; r < 8; ++r) {
                const int qr_ = wq + r + hi * 8;
                const int k0e = kb + lh, k1e = kb + 16 + lh;
                float s0 = (k0e <= qr_) ? st[0][r] * SCALE : -1e30f;
                float s1 = (k1e <= qr_) ? st[1][r] * SCALE : -1e30f;
                float rm = fmaxf(s0, s1);
                #pragma unroll
                for (int m = 8; m >= 1; m >>= 1) rm = fmaxf(rm, __shfl_xor(rm, m, 32));
                const float mn = fmaxf(m_run[r], rm);
                const float msafe = (mn < -1e29f) ? 0.f : mn;
                const float p0 = __expf(s0 - msafe);
                const float p1 = __expf(s1 - msafe);
                const float a_ = __expf(m_run[r] - msafe);
                float ps = p0 + p1;
                #pragma unroll
                for (int m = 8; m >= 1; m >>= 1) ps += __shfl_xor(ps, m, 32);
                l_run[r] = l_run[r] * a_ + ps;
                m_run[r] = mn;
                al[r] = a_;
                Ps[pbase + (r + hi * 8) * 40 + lh]      = f2bf(p0);
                Ps[pbase + (r + hi * 8) * 40 + 16 + lh] = f2bf(p1);
            }
            // V fragments load while P stores drain (DS is in-order per wave).
            Frag vf[8];
            #pragma unroll
            for (int t = 0; t < 8; ++t) {
                const u16* vp = &VT[buf][(t * 16 + lh) * 40 + hi * 8];
                vf[t].u4[0] = *(const uint4*)(vp);
                vf[t].u4[1] = *(const uint4*)(vp + 16);
            }
            #pragma unroll
            for (int t = 0; t < 8; ++t)
                #pragma unroll
                for (int r = 0; r < 8; ++r) o[t][r] *= al[r];

            Frag pf;
            pf.u4[0] = *(const uint4*)&Ps[pbase + lh * 40 + (hi ? 8 : 0)];
            pf.u4[1] = *(const uint4*)&Ps[pbase + lh * 40 + (hi ? 24 : 16)];
            #pragma unroll
            for (int t = 0; t < 8; ++t)
                o[t] = wmma_bf16(pf, vf[t], o[t]);
        }
        __syncthreads();
    }

    // ---- normalize and store bf16 ----
    float inv[8];
    #pragma unroll
    for (int r = 0; r < 8; ++r) inv[r] = 1.f / l_run[r];
    #pragma unroll
    for (int t = 0; t < 8; ++t) {
        const int col = hoff + t * 16 + lh;
        #pragma unroll
        for (int r = 0; r < 8; ++r) {
            const int row = wq + r + hi * 8;
            O[(size_t)row * ldo + col] = f2bf(o[t][r] * inv[r]);
        }
    }
}

// ---------------------------------------------------------------------------
// g = a1 * silu(a2), bf16 in/out, in-place into a1
// ---------------------------------------------------------------------------
__global__ __launch_bounds__(256) void silu_gate_k(u16* __restrict__ a1,
                                                   const u16* __restrict__ a2, size_t n) {
    size_t i = (size_t)blockIdx.x * 256 + threadIdx.x;
    if (i < n) {
        const float x1 = bf2f(a1[i]);
        const float x2 = bf2f(a2[i]);
        a1[i] = f2bf(x1 * (x2 / (1.f + __expf(-x2))));
    }
}

// ---------------------------------------------------------------------------
// host orchestration
// ---------------------------------------------------------------------------
extern "C" void kernel_launch(void* const* d_in, const int* in_sizes, int n_in,
                              void* d_out, int out_size, void* d_ws, size_t ws_size,
                              hipStream_t stream) {
    (void)in_sizes; (void)n_in; (void)out_size;
    const int S = 2048, H = 4096, NH = 32, FF = 11008, H3 = 12288;

    const float* hidden   = (const float*)d_in[0];
    const float* cosb     = (const float*)d_in[1];
    const float* sinb     = (const float*)d_in[2];
    // d_in[3]: attention_mask (all ones) -> causal only
    const float* ln1      = (const float*)d_in[4];
    const float* ln2      = (const float*)d_in[5];
    const float* c_attn_w = (const float*)d_in[6];
    const float* c_attn_b = (const float*)d_in[7];
    const float* c_proj_w = (const float*)d_in[8];
    const float* w1_w     = (const float*)d_in[9];
    const float* w2_w     = (const float*)d_in[10];
    const float* mlp_w    = (const float*)d_in[11];

    char* ws = (char*)d_ws;
    size_t off = 0;
    auto alloc = [&](size_t bytes) -> void* {
        size_t cur = (off + 255) & ~(size_t)255;
        off = cur + bytes;
        return (void*)(ws + cur);
    };
    // weights, stored transposed: WT[N][K] bf16
    u16*   wq_t = (u16*)alloc((size_t)H * H3 * 2);
    u16*   wp_t = (u16*)alloc((size_t)H * H  * 2);
    u16*   w1_t = (u16*)alloc((size_t)H * FF * 2);
    u16*   w2_t = (u16*)alloc((size_t)H * FF * 2);
    u16*   wm_t = (u16*)alloc((size_t)FF * H * 2);
    u16*   xn   = (u16*)alloc((size_t)S * H * 2);   // also reused as attn output
    u16*   qkvb = (u16*)alloc((size_t)S * H3 * 2);
    u16*   qr   = (u16*)alloc((size_t)S * H * 2);
    u16*   kr   = (u16*)alloc((size_t)S * H * 2);
    u16*   vtg  = (u16*)alloc((size_t)NH * 128 * S * 2);  // V^T per head
    float* hbuf = (float*)alloc((size_t)S * H * 4);
    u16*   yn   = (u16*)alloc((size_t)S * H * 2);
    u16*   a1   = (u16*)alloc((size_t)S * FF * 2);  // becomes g in-place
    u16*   a2   = (u16*)alloc((size_t)S * FF * 2);
    if (off > ws_size) return;   // workspace too small: nothing safe to do

    // transpose-cast weights: fp32 [K][N] -> bf16 [N][K]
    auto castT = [&](const float* src, u16* dst, int K, int N) {
        castT_k<<<dim3(K / 64, N / 64), 256, 0, stream>>>(src, N, dst, K);
    };
    castT(c_attn_w, wq_t, H, H3);
    castT(c_proj_w, wp_t, H, H);
    castT(w1_w,     w1_t, H, FF);
    castT(w2_w,     w2_t, H, FF);
    castT(mlp_w,    wm_t, FF, H);

    // x = rmsnorm(hidden, ln1)
    rmsnorm_k<<<S, 256, 0, stream>>>(hidden, ln1, xn, H);
    // qkv = x @ c_attn_w + b   (bf16 out)
    gemm_bf16_k<true, true, false><<<dim3(H3 / 128, S / 128), 256, 0, stream>>>(
        xn, wq_t, c_attn_b, nullptr, qkvb, S, H3, H);
    // RoPE -> qr, kr
    rope_k<<<(S * NH * 64) / 256, 256, 0, stream>>>(qkvb, cosb, sinb, qr, kr);
    // V^T per head: [S][128] (strided in qkv) -> [128][S]
    transpose16_k<<<dim3(S / 64, 2, NH), 256, 0, stream>>>(
        qkvb + 2 * H, H3, (size_t)128, vtg, S, (size_t)128 * S);
    // flash attention -> xn (reused)
    attn_k<<<dim3(S / 128, NH), 256, 0, stream>>>(qr, H, kr, H, vtg, S, xn, H);
    // h = hidden + attn @ c_proj_w   (fp32 out + residual)
    gemm_bf16_k<false, false, true><<<dim3(H / 128, S / 128), 256, 0, stream>>>(
        xn, wp_t, nullptr, hidden, hbuf, S, H, H);
    // y = rmsnorm(h, ln2)
    rmsnorm_k<<<S, 256, 0, stream>>>(hbuf, ln2, yn, H);
    // a1 = y @ w1 ; a2 = y @ w2   (bf16 out)
    gemm_bf16_k<true, false, false><<<dim3(FF / 128, S / 128), 256, 0, stream>>>(
        yn, w1_t, nullptr, nullptr, a1, S, FF, H);
    gemm_bf16_k<true, false, false><<<dim3(FF / 128, S / 128), 256, 0, stream>>>(
        yn, w2_t, nullptr, nullptr, a2, S, FF, H);
    // g = a1 * silu(a2) (in-place into a1)
    {
        size_t n = (size_t)S * FF;
        silu_gate_k<<<(unsigned)((n + 255) / 256), 256, 0, stream>>>(a1, a2, n);
    }
    // out = h + g @ mlp_proj_w   (fp32 out + residual) -> d_out
    gemm_bf16_k<false, false, true><<<dim3(H / 128, S / 128), 256, 0, stream>>>(
        a1, wm_t, nullptr, hbuf, (float*)d_out, S, H, FF);
}
